// CausalSelfAttention_90022514524569
// MI455X (gfx1250) — matrix operand order repaired and checked by reading, not compiled, not run on hardware
//
#include <hip/hip_runtime.h>
#include <hip/hip_bf16.h>

typedef __attribute__((ext_vector_type(16))) _Float16 v16h;
typedef __attribute__((ext_vector_type(8)))  float    v8f;
typedef __attribute__((ext_vector_type(2)))  _Float16 h2v;

#define EMBED 1024
#define NHEAD 16
#define HDIM  64
#define SEQ   2048
#define BATCH 2

// ---------------------------------------------------------------------------
// CDNA5 async global->LDS staging (ASYNCcnt path) + waits
// ---------------------------------------------------------------------------
__device__ __forceinline__ void async_b128(_Float16* ldst, const _Float16* gbase,
                                           unsigned byte_off) {
  unsigned laddr = (unsigned)(uintptr_t)ldst;  // flat LDS addr: low 32 bits = LDS offset
  asm volatile("global_load_async_to_lds_b128 %0, %1, %2"
               :: "v"(laddr), "v"(byte_off),
                  "s"((unsigned long long)(uintptr_t)gbase)
               : "memory");
}

__device__ __forceinline__ void wait_async() {
#if __has_builtin(__builtin_amdgcn_s_wait_asynccnt)
  __builtin_amdgcn_s_wait_asynccnt(0);
#else
  asm volatile("s_wait_asynccnt 0" ::: "memory");
#endif
}

// ---------------------------------------------------------------------------
// WMMA helpers (gfx1250 wave32). Fragment layouts per ISA 7.12.2.
// ---------------------------------------------------------------------------
__device__ __forceinline__ v8f wmma16(v16h a, v16h b, v8f c) {
  return __builtin_amdgcn_wmma_f32_16x16x32_f16(
      false, a, false, b, (short)0, c, false, false);
}

// A fragment: 16x32 f16, row-major source, stride ld halves.
__device__ __forceinline__ v16h frag_a_ld(const _Float16* base, int ld) {
  int lane = threadIdx.x & 31;
  int m = lane & 15, hl = lane >> 4;
  v16h f;
#pragma unroll
  for (int e = 0; e < 8; ++e) f[e] = base[m * ld + 8 * hl + e];
#pragma unroll
  for (int e = 0; e < 8; ++e) f[8 + e] = base[m * ld + 16 + 8 * hl + e];
  return f;
}

// B fragment: 32x16 f16, source stored TRANSPOSED: base[n*ld + k].
__device__ __forceinline__ v16h frag_b_tr(const _Float16* base, int ld) {
  int lane = threadIdx.x & 31;
  int n = lane & 15, hl = lane >> 4;
  v16h f;
#pragma unroll
  for (int e = 0; e < 16; ++e) f[e] = base[n * ld + ((e < 8) ? 8 * hl + e : 16 + 8 * hl + (e - 8))];
  return f;
}

// ---------------------------------------------------------------------------
// f32 -> f16 conversion
// ---------------------------------------------------------------------------
__global__ __launch_bounds__(256) void cvt_f32_f16(const float* __restrict__ s,
                                                   _Float16* __restrict__ d, int n) {
  int i = blockIdx.x * 256 + threadIdx.x;
  if (i < n) d[i] = (_Float16)s[i];
}

// ---------------------------------------------------------------------------
// GEMM: C[M,N] = A[M,K] f16 * B[K,N] f16 + bias[N]
// Block 256 thr = 8 waves; block tile 128(M) x 128(N); wave tile 32x64 (2x4 wmma).
// Double-buffered LDS; A tile staged via async-to-LDS, B tile transposed via VGPRs.
// mode 0: scatter into Q/K/V f16 [B,H,T,D].  mode 1: f32 out[M,N] + bias.
// ---------------------------------------------------------------------------
__global__ __launch_bounds__(256) void gemm_wmma(
    const _Float16* __restrict__ A, const _Float16* __restrict__ Bm,
    const float* __restrict__ bias, int N, int K, int mode,
    _Float16* __restrict__ qh, _Float16* __restrict__ kh, _Float16* __restrict__ vh,
    float* __restrict__ outf) {
  __shared__ alignas(16) _Float16 Alds[2][128 * 32];  // [m][k], ld=32
  __shared__ alignas(16) _Float16 Blds[2][128 * 32];  // transposed [n][k], ld=32

  const int m0 = blockIdx.y * 128;
  const int n0 = blockIdx.x * 128;
  const int t = threadIdx.x;
  const int w = t >> 5;
  const int wm = (w & 3) * 32;    // 4 waves down M
  const int wn = (w >> 2) * 64;   // 2 waves across N
  const int lane = t & 31, n_l = lane & 15, hl = lane >> 4;

  // staging indices (computed once)
  const int a_row = t >> 1, a_seg = (t & 1) * 16;         // A: 128 rows x 2 chunks
  const int b_k = (t >> 4) * 2, b_n = (t & 15) * 8;       // B: 16 k-pairs x 16 n-segs

  v8f acc[2][4];
#pragma unroll
  for (int i = 0; i < 2; ++i)
#pragma unroll
    for (int j = 0; j < 4; ++j) acc[i][j] = (v8f){};

  auto stage = [&](int buf, int kk) {
    // A tile 128x32: async b128 per thread
    async_b128(&Alds[buf][a_row * 32 + a_seg], A,
               (unsigned)(((m0 + a_row) * K + kk + a_seg) * 2));
    // B tile 32x128 -> transposed [n][k]; read 2 k-rows, store 8x b32 pairs
    alignas(16) _Float16 t0[8], t1[8];
    *(int4*)t0 = *(const int4*)(Bm + (size_t)(kk + b_k) * N + n0 + b_n);
    *(int4*)t1 = *(const int4*)(Bm + (size_t)(kk + b_k + 1) * N + n0 + b_n);
#pragma unroll
    for (int j = 0; j < 8; ++j)
      *(h2v*)&Blds[buf][(b_n + j) * 32 + b_k] = (h2v){t0[j], t1[j]};
  };

  auto compute = [&](int buf) {
    v16h af[2], bf[4];
#pragma unroll
    for (int i = 0; i < 2; ++i) af[i] = frag_a_ld(&Alds[buf][(wm + i * 16) * 32], 32);
#pragma unroll
    for (int j = 0; j < 4; ++j) bf[j] = frag_b_tr(&Blds[buf][(wn + j * 16) * 32], 32);
#pragma unroll
    for (int i = 0; i < 2; ++i)
#pragma unroll
      for (int j = 0; j < 4; ++j) acc[i][j] = wmma16(af[i], bf[j], acc[i][j]);
  };

  stage(0, 0);
  int buf = 0;
  for (int kk = 32; kk < K; kk += 32) {
    wait_async();
    __syncthreads();            // buf fully staged for all waves
    stage(buf ^ 1, kk);         // overlap next stage with compute
    compute(buf);
    buf ^= 1;
  }
  wait_async();
  __syncthreads();
  compute(buf);

  // ---- epilogue: C/D layout m = el + 8*hl, n = lane&15
#pragma unroll
  for (int j = 0; j < 4; ++j) {
    int gn = n0 + wn + j * 16 + n_l;
    float bv = bias[gn];
#pragma unroll
    for (int i = 0; i < 2; ++i) {
#pragma unroll
      for (int el = 0; el < 8; ++el) {
        float vv = acc[i][j][el] + bv;
        int gm = m0 + wm + i * 16 + el + 8 * hl;
        if (mode == 0) {
          int bb = gm >> 11, tt = gm & (SEQ - 1);
          int s = gn >> 10, e = gn & (EMBED - 1);
          int hh = e >> 6, d = e & (HDIM - 1);
          size_t idx = ((size_t)((bb * NHEAD + hh) * SEQ + tt)) * HDIM + d;
          _Float16* dst = (s == 0) ? qh : (s == 1) ? kh : vh;
          dst[idx] = (_Float16)vv;
        } else {
          outf[(size_t)gm * N + gn] = vv;
        }
      }
    }
  }
}

// ---------------------------------------------------------------------------
// Attention tile loaders (block = 128 threads)
// ---------------------------------------------------------------------------
__device__ __forceinline__ void load_tile_rm_async(_Float16* dst, const _Float16* gbase,
                                                   unsigned elem_off) {
  // 64x64 f16 row-major: 256 chunks of 16B, 2 per thread
  int t = threadIdx.x;
#pragma unroll
  for (int i = 0; i < 2; ++i) {
    int idx = t + 128 * i;
    int r = idx >> 2, seg = (idx & 3) * 16;
    async_b128(dst + r * 64 + seg, gbase, (elem_off + r * 64 + seg) * 2u);
  }
}

__device__ __forceinline__ void load_tile_tr(_Float16* dst, const _Float16* src) {
  // dst[d*64 + r] = src[r*64 + d]  (transpose during staging)
  int t = threadIdx.x;
#pragma unroll
  for (int i = 0; i < 4; ++i) {
    int idx = t + 128 * i;
    int r = idx >> 3, d0 = (idx & 7) * 8;
    alignas(16) _Float16 tmp[8];
    *(int4*)tmp = *(const int4*)(src + (size_t)r * 64 + d0);
#pragma unroll
    for (int j = 0; j < 8; ++j) dst[(d0 + j) * 64 + r] = tmp[j];
  }
}

// ---------------------------------------------------------------------------
// Flash attention. Grid (T/64, B*H), block 128 (4 waves x 16 q-rows).
// q/k/v f16 [B,H,T,D]; o f16 [B,T,E].
// ---------------------------------------------------------------------------
__global__ __launch_bounds__(128) void attn_fwd(
    const _Float16* __restrict__ q, const _Float16* __restrict__ k,
    const _Float16* __restrict__ v, _Float16* __restrict__ o) {
  __shared__ alignas(16) _Float16 Qlds[64 * 64];
  __shared__ alignas(16) _Float16 Klds[64 * 64];     // [key][dim]
  __shared__ alignas(16) _Float16 Vlds[64 * 64];     // transposed [dim][key]
  __shared__ alignas(16) _Float16 Plds[4][16 * 64];  // per-wave P staging

  const int bh = blockIdx.y;
  const int q0 = blockIdx.x * 64;
  const unsigned base = (unsigned)bh * SEQ * HDIM;
  const int t = threadIdx.x;
  const int w = t >> 5, lane = t & 31, n_l = lane & 15, hl = lane >> 4;

  load_tile_rm_async(Qlds, q, base + (unsigned)q0 * HDIM);
  wait_async();
  __syncthreads();
  v16h qa[2];
  qa[0] = frag_a_ld(Qlds + (w * 16) * 64, 64);
  qa[1] = frag_a_ld(Qlds + (w * 16) * 64 + 32, 64);

  float m_i[8], l_i[8];
  v8f O[4];
#pragma unroll
  for (int el = 0; el < 8; ++el) { m_i[el] = -__builtin_inff(); l_i[el] = 0.f; }
#pragma unroll
  for (int c = 0; c < 4; ++c) O[c] = (v8f){};

  const int nkt = q0 / 64 + 1;
  for (int kt = 0; kt < nkt; ++kt) {
    __syncthreads();  // all waves done reading previous K/V
    load_tile_rm_async(Klds, k, base + (unsigned)kt * 64 * HDIM);
    load_tile_tr(Vlds, v + base + (size_t)kt * 64 * HDIM);
    wait_async();
    __syncthreads();

    // ---- S = Q * K^T
    v8f S[4];
#pragma unroll
    for (int c = 0; c < 4; ++c) {
      S[c] = (v8f){};
#pragma unroll
      for (int kc = 0; kc < 2; ++kc) {
        v16h bf = frag_b_tr(Klds + (c * 16) * 64 + kc * 32, 64);
        S[c] = wmma16(qa[kc], bf, S[c]);
      }
    }

    // ---- scale + causal mask + online softmax
    const bool diag = (kt == nkt - 1);
    float mn[8];
#pragma unroll
    for (int el = 0; el < 8; ++el) mn[el] = m_i[el];
#pragma unroll
    for (int c = 0; c < 4; ++c) {
#pragma unroll
      for (int el = 0; el < 8; ++el) {
        float s = S[c][el] * 0.125f;  // 1/sqrt(64)
        if (diag) {
          int col = kt * 64 + c * 16 + n_l;
          int row = q0 + w * 16 + el + 8 * hl;
          if (col > row) s = -__builtin_inff();
        }
        S[c][el] = s;
        mn[el] = fmaxf(mn[el], s);
      }
    }
#pragma unroll
    for (int el = 0; el < 8; ++el)
#pragma unroll
      for (int off = 1; off < 16; off <<= 1)
        mn[el] = fmaxf(mn[el], __shfl_xor(mn[el], off, 32));

    float alpha[8], rs[8];
#pragma unroll
    for (int el = 0; el < 8; ++el) {
      alpha[el] = __expf(m_i[el] - mn[el]);
      m_i[el] = mn[el];
      rs[el] = 0.f;
    }
#pragma unroll
    for (int c = 0; c < 4; ++c)
#pragma unroll
      for (int el = 0; el < 8; ++el) {
        float p = __expf(S[c][el] - mn[el]);
        S[c][el] = p;
        rs[el] += p;
      }
#pragma unroll
    for (int el = 0; el < 8; ++el)
#pragma unroll
      for (int off = 1; off < 16; off <<= 1)
        rs[el] += __shfl_xor(rs[el], off, 32);
#pragma unroll
    for (int el = 0; el < 8; ++el) l_i[el] = l_i[el] * alpha[el] + rs[el];
#pragma unroll
    for (int c = 0; c < 4; ++c)
#pragma unroll
      for (int el = 0; el < 8; ++el) O[c][el] *= alpha[el];

    // ---- P through per-wave LDS, back in as A fragment
    _Float16* Pw = &Plds[w][0];
#pragma unroll
    for (int c = 0; c < 4; ++c)
#pragma unroll
      for (int el = 0; el < 8; ++el)
        Pw[(el + 8 * hl) * 64 + c * 16 + n_l] = (_Float16)S[c][el];
    asm volatile("s_wait_dscnt 0" ::: "memory");

    v16h pa[2];
    pa[0] = frag_a_ld(Pw, 64);
    pa[1] = frag_a_ld(Pw + 32, 64);
#pragma unroll
    for (int c = 0; c < 4; ++c) {
#pragma unroll
      for (int kc = 0; kc < 2; ++kc) {
        v16h vf = frag_b_tr(Vlds + (c * 16) * 64 + kc * 32, 64);
        O[c] = wmma16(pa[kc], vf, O[c]);
      }
    }
  }

  // ---- epilogue: normalize, merge heads into [B,T,E] f16
  const int b = bh >> 4, head = bh & (NHEAD - 1);
#pragma unroll
  for (int c = 0; c < 4; ++c)
#pragma unroll
    for (int el = 0; el < 8; ++el) {
      float ov = O[c][el] / l_i[el];
      int tt = q0 + w * 16 + el + 8 * hl;
      int e = head * HDIM + c * 16 + n_l;
      o[((size_t)(b * SEQ + tt)) * EMBED + e] = (_Float16)ov;
    }
}

// ---------------------------------------------------------------------------
// Launch
// ---------------------------------------------------------------------------
extern "C" void kernel_launch(void* const* d_in, const int* in_sizes, int n_in,
                              void* d_out, int out_size, void* d_ws, size_t ws_size,
                              hipStream_t stream) {
  const float* x    = (const float*)d_in[0];
  const float* Wqkv = (const float*)d_in[1];
  const float* bqkv = (const float*)d_in[2];
  const float* Wout = (const float*)d_in[3];
  const float* bout = (const float*)d_in[4];
  float* out = (float*)d_out;

  char* ws = (char*)d_ws;
  size_t off = 0;
  auto alloc = [&](size_t bytes) -> char* {
    char* p = ws + off;
    off += (bytes + 255) & ~(size_t)255;
    return p;
  };
  const size_t nBTE = (size_t)BATCH * SEQ * EMBED;
  _Float16* x_h    = (_Float16*)alloc(nBTE * 2);
  _Float16* wqkv_h = (_Float16*)alloc((size_t)EMBED * 3 * EMBED * 2);
  _Float16* wout_h = (_Float16*)alloc((size_t)EMBED * EMBED * 2);
  _Float16* q_h    = (_Float16*)alloc(nBTE * 2);
  _Float16* k_h    = (_Float16*)alloc(nBTE * 2);
  _Float16* v_h    = (_Float16*)alloc(nBTE * 2);
  _Float16* att_h  = (_Float16*)alloc(nBTE * 2);

  int nx = (int)nBTE;
  int nw1 = EMBED * 3 * EMBED;
  int nw2 = EMBED * EMBED;
  cvt_f32_f16<<<(nx + 255) / 256, 256, 0, stream>>>(x, x_h, nx);
  cvt_f32_f16<<<(nw1 + 255) / 256, 256, 0, stream>>>(Wqkv, wqkv_h, nw1);
  cvt_f32_f16<<<(nw2 + 255) / 256, 256, 0, stream>>>(Wout, wout_h, nw2);

  // QKV projection: [4096,1024] x [1024,3072] -> Q/K/V [B,H,T,D] f16
  dim3 g1(3 * EMBED / 128, BATCH * SEQ / 128);
  gemm_wmma<<<g1, 256, 0, stream>>>(x_h, wqkv_h, bqkv, 3 * EMBED, EMBED, 0,
                                    q_h, k_h, v_h, nullptr);

  // Flash attention
  dim3 g2(SEQ / 64, BATCH * NHEAD);
  attn_fwd<<<g2, 128, 0, stream>>>(q_h, k_h, v_h, att_h);

  // Output projection: [4096,1024] x [1024,1024] -> f32 out
  dim3 g3(EMBED / 128, BATCH * SEQ / 128);
  gemm_wmma<<<g3, 256, 0, stream>>>(att_h, wout_h, bout, EMBED, EMBED, 1,
                                    nullptr, nullptr, nullptr, out);
}
